// VQSign_14018773254241
// MI455X (gfx1250) — matrix operand order repaired and checked by reading, not compiled, not run on hardware
//
#include <hip/hip_runtime.h>
#include <hip/hip_bf16.h>
#include <stdint.h>

// ---------------------------------------------------------------------------
// VQ-CPC forward for MI455X (gfx1250, wave32, WMMA bf16 + async-to-LDS).
// B=64, T=512, d=1024, Kc=256, N=B*T=32768, 3d=3072.
// GEMMs: v_wmma_f32_16x16x32_bf16 (bf16 operands, f32 accumulate).
// A-panels staged in LDS via global_load_async_to_lds_b128 (ASYNCcnt path);
// GRU scan streams h out via global_store_async_from_lds_b128.
// ---------------------------------------------------------------------------

typedef __bf16 bf16;
typedef __attribute__((ext_vector_type(16))) bf16  v16bf;
typedef __attribute__((ext_vector_type(8)))  bf16  v8bf;
typedef __attribute__((ext_vector_type(4)))  bf16  v4bf;
typedef __attribute__((ext_vector_type(8)))  float v8f;

#define Bsz 64
#define Tsz 512
#define Dsz 1024
#define Kc  256
#define Nrows (Bsz * Tsz)          // 32768
#define D3  (3 * Dsz)              // 3072

__device__ __forceinline__ v8f wmma_bf16(v16bf a, v16bf b, v8f c) {
  return __builtin_amdgcn_wmma_f32_16x16x32_bf16(false, a, false, b, (short)0, c,
                                                 false, false);
}

// Low 32 bits of a generic pointer into the LDS aperture == LDS byte offset.
__device__ __forceinline__ unsigned lds_addr32(const void* p) {
  return (unsigned)(uintptr_t)p;
}

// gfx1250 async DMA lane ops (tracked by ASYNCcnt).
__device__ __forceinline__ void async_ld_b128(unsigned ldsOff, const void* g) {
  asm volatile("global_load_async_to_lds_b128 %0, %1, off"
               :: "v"(ldsOff), "v"((unsigned long long)(uintptr_t)g) : "memory");
}
__device__ __forceinline__ void async_st_b128(const void* g, unsigned ldsOff) {
  asm volatile("global_store_async_from_lds_b128 %0, %1, off"
               :: "v"((unsigned long long)(uintptr_t)g), "v"(ldsOff) : "memory");
}
__device__ __forceinline__ void wait_async0() {
  asm volatile("s_wait_asynccnt 0x0" ::: "memory");
}

// A fragment (16x32 bf16): lane holds row M=lane%16; VGPR0-7 hold
// K=(lane/16)*16..+15 (contiguous, matches B). p = row + k0 + (lane/16)*16.
__device__ __forceinline__ v16bf make_afrag(const bf16* p) {
  v8bf lo = *(const v8bf*)p;
  v8bf hi = *(const v8bf*)(p + 8);
  return __builtin_shufflevector(lo, hi, 0, 1, 2, 3, 4, 5, 6, 7,
                                 8, 9, 10, 11, 12, 13, 14, 15);
}

__device__ __forceinline__ float sigm(float x) { return 1.0f / (1.0f + expf(-x)); }

__device__ __forceinline__ unsigned orderedKey(float s) {
  unsigned u = __float_as_uint(s);
  return (u & 0x80000000u) ? ~u : (u | 0x80000000u);
}

__device__ float blockReduceSum(float v) {
  __shared__ float red[8];
  const int lane = threadIdx.x & 31, wid = threadIdx.x >> 5;
#pragma unroll
  for (int o = 16; o > 0; o >>= 1) v += __shfl_xor(v, o, 32);
  __syncthreads();
  if (lane == 0) red[wid] = v;
  __syncthreads();
  float r = 0.0f;
  if (threadIdx.x < (int)(blockDim.x >> 5)) r = red[threadIdx.x];
  if (wid == 0) {
#pragma unroll
    for (int o = 4; o > 0; o >>= 1) r += __shfl_xor(r, o, 32);
  }
  return r;  // valid on thread 0
}

// --------------------------- elementwise converts --------------------------

__global__ void cvt_f32_bf16(const float* __restrict__ in, bf16* __restrict__ out,
                             long n4) {
  long i = (long)blockIdx.x * blockDim.x + threadIdx.x;
  if (i < n4) {
    float4 v = ((const float4*)in)[i];
    v4bf o;
    o[0] = (bf16)v.x; o[1] = (bf16)v.y; o[2] = (bf16)v.z; o[3] = (bf16)v.w;
    ((v4bf*)out)[i] = o;
  }
}

__global__ void codebook_prep(const float* __restrict__ cb, bf16* __restrict__ Ch,
                              float* __restrict__ cnorm) {
  const int row = blockIdx.x;
  float s = 0.0f;
  for (int e = threadIdx.x; e < Dsz; e += blockDim.x) {
    float v = cb[(size_t)row * Dsz + e];
    Ch[(size_t)row * Dsz + e] = (bf16)v;
    s += v * v;
  }
  float tot = blockReduceSum(s);
  if (threadIdx.x == 0) cnorm[row] = tot;
}

// ----------------------------- quantize (WMMA) -----------------------------
// Block = 16 rows of Z vs all 256 codes; A-panel (32 KB) async-staged in LDS.
__global__ void quantize_kernel(const float* __restrict__ Z, const float* __restrict__ cb,
                                const bf16* __restrict__ Zh, const bf16* __restrict__ Ch,
                                const float* __restrict__ cnorm,
                                float* __restrict__ Qout, float* __restrict__ idxF,
                                int* __restrict__ idxI, float* __restrict__ mseAcc) {
  const int lane = threadIdx.x & 31, wid = threadIdx.x >> 5;
  const int col = lane & 15, hf = lane >> 4;
  const int m0 = blockIdx.x * 16;

  __shared__ bf16 As[16 * Dsz];              // 32 KB A panel
  __shared__ unsigned long long best[16];
  if (threadIdx.x < 16) best[threadIdx.x] = ~0ull;

  {
    const unsigned abase = lds_addr32(As);
    const bf16* gsrc = Zh + (size_t)m0 * Dsz;
#pragma unroll
    for (int i = 0; i < (16 * Dsz / 8) / 256; ++i) {
      int c = threadIdx.x + i * 256;
      async_ld_b128(abase + (unsigned)c * 16, gsrc + (size_t)c * 8);
    }
    wait_async0();
  }
  __syncthreads();

  v8f acc[2] = {};
  const bf16* arow = As + col * Dsz + hf * 16;
  for (int k0 = 0; k0 < Dsz; k0 += 32) {
    v16bf a = make_afrag(arow + k0);         // ds_load_b128 x2
#pragma unroll
    for (int j = 0; j < 2; ++j) {
      const int code = wid * 32 + j * 16 + col;
      v16bf b = *(const v16bf*)(Ch + (size_t)code * Dsz + k0 + hf * 16);
      acc[j] = wmma_bf16(a, b, acc[j]);
    }
  }
#pragma unroll
  for (int j = 0; j < 2; ++j) {
    const int code = wid * 32 + j * 16 + col;
    const float cn = cnorm[code];
#pragma unroll
    for (int r = 0; r < 8; ++r) {
      float s = cn - 2.0f * acc[j][r];
      unsigned long long key =
          ((unsigned long long)orderedKey(s) << 32) | (unsigned)code;
      atomicMin(&best[hf * 8 + r], key);     // ds_min_u64
    }
  }
  __syncthreads();

  float msel = 0.0f;
  for (int lr = 0; lr < 16; ++lr) {
    const int row = m0 + lr;
    const int code = (int)(best[lr] & 0xffffffffu);
    if (threadIdx.x == 0) { idxF[row] = (float)code; idxI[row] = code; }
    for (int e = threadIdx.x; e < Dsz; e += blockDim.x) {
      float q = cb[(size_t)code * Dsz + e];
      Qout[(size_t)row * Dsz + e] = q;
      float dd = Z[(size_t)row * Dsz + e] - q;
      msel += dd * dd;
    }
  }
  float tot = blockReduceSum(msel);
  if (threadIdx.x == 0) atomicAdd(mseAcc, tot);  // global_atomic_add_f32
}

// --------------------- generic C = A @ W^T + bias (WMMA) --------------------
// A: MxK bf16 row-major (K must be 1024 here); W: NxK bf16 row-major.
// Block = 8 waves covering 16x512; shared 16xK A panel async-staged in LDS.
template <typename OutT>
__global__ void gemm_abt(const bf16* __restrict__ A, const bf16* __restrict__ W,
                         const float* __restrict__ bias, OutT* __restrict__ C,
                         int M, int N, int K) {
  const int lane = threadIdx.x & 31, wid = threadIdx.x >> 5;
  const int col = lane & 15, hf = lane >> 4;
  const int m0 = blockIdx.x * 16;
  const int n0 = blockIdx.y * 512 + wid * 64;

  __shared__ bf16 As[16 * Dsz];              // 32 KB
  {
    const unsigned abase = lds_addr32(As);
    const bf16* gsrc = A + (size_t)m0 * K;
    const int chunks = 16 * K / 8;
    for (int c = threadIdx.x; c < chunks; c += blockDim.x)
      async_ld_b128(abase + (unsigned)c * 16, gsrc + (size_t)c * 8);
    wait_async0();
  }
  __syncthreads();

  v8f acc[4] = {};
  const bf16* arow = As + col * K + hf * 16;
  for (int k0 = 0; k0 < K; k0 += 32) {
    v16bf a = make_afrag(arow + k0);
    __builtin_prefetch(W + (size_t)(n0 + col) * K + k0 + 512, 0, 1);
#pragma unroll
    for (int j = 0; j < 4; ++j) {
      const bf16* wp = W + (size_t)(n0 + j * 16 + col) * K + k0 + hf * 16;
      v16bf b = *(const v16bf*)wp;
      acc[j] = wmma_bf16(a, b, acc[j]);
    }
  }
#pragma unroll
  for (int j = 0; j < 4; ++j) {
    const int n = n0 + j * 16 + col;
    const float bs = bias ? bias[n] : 0.0f;
#pragma unroll
    for (int r = 0; r < 8; ++r) {
      const int m = m0 + hf * 8 + r;
      C[(size_t)m * N + n] = (OutT)(acc[j][r] + bs);
    }
  }
}

// ------------------------------ grid barrier --------------------------------
__device__ void grid_sync(unsigned* cnt, unsigned* gen, unsigned nb) {
  __syncthreads();
  if (threadIdx.x == 0) {
    unsigned g = __hip_atomic_load(gen, __ATOMIC_ACQUIRE, __HIP_MEMORY_SCOPE_AGENT);
    if (__hip_atomic_fetch_add(cnt, 1u, __ATOMIC_ACQ_REL, __HIP_MEMORY_SCOPE_AGENT) ==
        nb - 1u) {
      __hip_atomic_store(cnt, 0u, __ATOMIC_RELAXED, __HIP_MEMORY_SCOPE_AGENT);
      __hip_atomic_fetch_add(gen, 1u, __ATOMIC_ACQ_REL, __HIP_MEMORY_SCOPE_AGENT);
    } else {
      while (__hip_atomic_load(gen, __ATOMIC_ACQUIRE, __HIP_MEMORY_SCOPE_AGENT) == g)
        __builtin_amdgcn_s_sleep(1);
    }
  }
  __syncthreads();
}

// ------------------------- persistent GRU scan (WMMA) -----------------------
// 8 blocks x 256 threads = 64 waves; wave wg owns hidden slice [wg*16,+16).
// Per step: async-stage full h_old (128 KB) into LDS once per block; 12 WMMA
// tiles (4 M x 3 gates) x 32 K-steps per wave; results go through a 64x128 LDS
// tile and stream out with coalesced global_store_async_from_lds_b128 (both
// the h double-buffer and the Hh history). Dynamic LDS: 128 KB + 16 KB.
__global__ void gru_scan(const float* __restrict__ GiC, const int* __restrict__ idx,
                         const bf16* __restrict__ Whh, const float* __restrict__ bhh,
                         bf16* hb0, bf16* hb1, bf16* __restrict__ Hh,
                         unsigned* cnt, unsigned* gen) {
  extern __shared__ bf16 smem[];
  bf16* hs   = smem;                  // [64][1024] staged h_old
  bf16* tile = smem + Bsz * Dsz;      // [64][128] output slice of this block

  const int lane = threadIdx.x & 31, wid = threadIdx.x >> 5;
  const int wg = blockIdx.x * (blockDim.x >> 5) + wid;  // 0..63
  const int col = lane & 15, hf = lane >> 4;
  const int j0 = wg * 16;
  const int j = j0 + col;
  const int jb = blockIdx.x * 128;                      // block's hidden base
  const float brv = bhh[j], bzv = bhh[Dsz + j], bnv = bhh[2 * Dsz + j];
  bf16* bufs[2] = {hb0, hb1};
  const unsigned hsBase = lds_addr32(hs);
  const unsigned tileBase = lds_addr32(tile);

  for (int t = 0; t < Tsz; ++t) {
    const bf16* hOld = bufs[t & 1];
    bf16* hNew = bufs[(t + 1) & 1];

    // stage h_old: 128 KB = 8192 b128 chunks, 32 per thread
#pragma unroll
    for (int i = 0; i < 32; ++i) {
      int c = threadIdx.x + i * 256;
      async_ld_b128(hsBase + (unsigned)c * 16, hOld + (size_t)c * 8);
    }
    wait_async0();
    __syncthreads();

    v8f acc[4][3] = {};
    for (int k0 = 0; k0 < Dsz; k0 += 32) {
      v16bf bf[3];
#pragma unroll
      for (int g = 0; g < 3; ++g)
        bf[g] = *(const v16bf*)(Whh + (size_t)(g * Dsz + j0 + col) * Dsz + k0 + hf * 16);
#pragma unroll
      for (int m = 0; m < 4; ++m) {
        v16bf a = make_afrag(hs + (m * 16 + col) * Dsz + k0 + hf * 16);  // ds_load
#pragma unroll
        for (int g = 0; g < 3; ++g) acc[m][g] = wmma_bf16(a, bf[g], acc[m][g]);
      }
    }

#pragma unroll
    for (int m = 0; m < 4; ++m) {
#pragma unroll
      for (int r = 0; r < 8; ++r) {
        const int b = m * 16 + hf * 8 + r;
        const int ci = idx[b * Tsz + t];
        const float* gi = GiC + (size_t)ci * D3;
        float rr = sigm(gi[j] + acc[m][0][r] + brv);
        float zz = sigm(gi[Dsz + j] + acc[m][1][r] + bzv);
        float nn = tanhf(gi[2 * Dsz + j] + rr * (acc[m][2][r] + bnv));
        float hp = (float)hs[b * Dsz + j];
        float hv = (1.0f - zz) * nn + zz * hp;
        tile[b * 128 + wid * 16 + col] = (bf16)hv;     // ds_store_b16
      }
    }
    __syncthreads();

    // stream tile out: 1024 b128 chunks -> h double-buffer + Hh history
#pragma unroll
    for (int i = 0; i < 4; ++i) {
      int c = threadIdx.x + i * 256;                   // 0..1023
      int b = c >> 4, sub = c & 15;
      unsigned l = tileBase + (unsigned)(b * 128 + sub * 8) * 2;
      async_st_b128(hNew + (size_t)b * Dsz + jb + sub * 8, l);
      async_st_b128(Hh + ((size_t)b * Tsz + t) * Dsz + jb + sub * 8, l);
    }
    wait_async0();
    grid_sync(cnt, gen, gridDim.x);
  }
}

// --------------------------- CPC context loss (WMMA) ------------------------
__global__ void ctx_loss(const bf16* __restrict__ Hp, const bf16* __restrict__ Zh,
                         float* posAcc, float* negAcc) {
  const int p = blockIdx.x;
  int k, t;
  if (p < 511) { k = 1; t = p; }
  else if (p < 1021) { k = 2; t = p - 511; }
  else { k = 3; t = p - 1021; }

  const int lane = threadIdx.x & 31, wid = threadIdx.x >> 5;
  const int col = lane & 15, hf = lane >> 4;
  const int m = wid & 3, np = wid >> 2;

  const bf16* arow = Hp + ((size_t)(m * 16 + col) * Tsz + t) * Dsz + hf * 16;
  v8f acc[2] = {};
  for (int k0 = 0; k0 < Dsz; k0 += 32) {
    v16bf a = make_afrag(arow + k0);
#pragma unroll
    for (int jj = 0; jj < 2; ++jj) {
      const int c = np * 32 + jj * 16 + col;
      v16bf b = *(const v16bf*)(Zh + ((size_t)c * Tsz + t + k) * Dsz + k0 + hf * 16);
      acc[jj] = wmma_bf16(a, b, acc[jj]);
    }
  }

  float ps = 0.0f, ns = 0.0f;
#pragma unroll
  for (int jj = 0; jj < 2; ++jj) {
    const int c = np * 32 + jj * 16 + col;
#pragma unroll
    for (int r = 0; r < 8; ++r) {
      const float s = acc[jj][r];
      const int b = m * 16 + hf * 8 + r;
      ns += -logf(sigm(-s) + 1e-8f);              // 1-sigmoid(s) == sigmoid(-s)
      if (b == c) ps += -logf(sigm(s) + 1e-8f);
    }
  }
  ps = blockReduceSum(ps);
  ns = blockReduceSum(ns);
  if (threadIdx.x == 0) { atomicAdd(posAcc, ps); atomicAdd(negAcc, ns); }
}

__global__ void finalize(const float* __restrict__ sc, float* __restrict__ outS) {
  if (threadIdx.x == 0) {
    const float mse = sc[0], pos = sc[1], neg = sc[2];
    const float lvq = 1.25f * mse / (float)((size_t)Nrows * Dsz);  // (1+GAMMA)*MSE
    const float lcp = (pos / (float)Bsz + 0.1f * neg / (float)(Bsz * Bsz)) /
                      (3.0f * (float)(Tsz - 3));
    outS[0] = lcp + lvq; outS[1] = lcp; outS[2] = lvq;
  }
}

// ------------------------------- launcher -----------------------------------
extern "C" void kernel_launch(void* const* d_in, const int* in_sizes, int n_in,
                              void* d_out, int out_size, void* d_ws, size_t ws_size,
                              hipStream_t stream) {
  (void)in_sizes; (void)n_in; (void)out_size; (void)ws_size;
  const float* Z    = (const float*)d_in[0];
  const float* cb   = (const float*)d_in[1];
  const float* Wih  = (const float*)d_in[2];
  const float* Whh  = (const float*)d_in[3];
  const float* bih  = (const float*)d_in[4];
  const float* bhh  = (const float*)d_in[5];
  const float* Pw   = (const float*)d_in[6];
  const float* pb   = (const float*)d_in[7];
  float* out = (float*)d_out;
  float* Qout = out;                         // 33554432
  float* idxF = out + (size_t)Nrows * Dsz;   // 32768
  float* outS = idxF + Nrows;                // 3 scalars

  // workspace layout (~221 MB total)
  char* w = (char*)d_ws;
  float*    sc    = (float*)w;               // [0]=mse [1]=pos [2]=neg
  unsigned* cnt   = (unsigned*)(w + 16);
  unsigned* gen   = (unsigned*)(w + 20);
  size_t off = 256;
  bf16* hb0   = (bf16*)(w + off); off += (size_t)Bsz * Dsz * 2;        // 128 KB
  bf16* hb1   = (bf16*)(w + off); off += (size_t)Bsz * Dsz * 2;        // 128 KB
  float* cn   = (float*)(w + off); off += Kc * 4;
  int*  idxI  = (int*)(w + off);   off += (size_t)Nrows * 4;           // 128 KB
  bf16* Ch    = (bf16*)(w + off);  off += (size_t)Kc * Dsz * 2;        // 0.5 MB
  bf16* WihH  = (bf16*)(w + off);  off += (size_t)D3 * Dsz * 2;        // 6 MB
  bf16* WhhH  = (bf16*)(w + off);  off += (size_t)D3 * Dsz * 2;        // 6 MB
  bf16* PwH   = (bf16*)(w + off);  off += (size_t)Dsz * Dsz * 2;       // 2 MB
  float* GiC  = (float*)(w + off); off += (size_t)Kc * D3 * 4;         // 3 MB
  bf16* Zh    = (bf16*)(w + off);  off += (size_t)Nrows * Dsz * 2;     // 64 MB
  bf16* Hh    = (bf16*)(w + off);  off += (size_t)Nrows * Dsz * 2;     // 64 MB
  bf16* Hp    = (bf16*)(w + off);  off += (size_t)Nrows * Dsz * 2;     // 64 MB

  // zero loss accumulators, barrier counters and h0 (bf16 zero == 0x0000)
  hipMemsetAsync(d_ws, 0, 256 + 2 * (size_t)Bsz * Dsz * 2, stream);

  long n4;
  n4 = (long)Nrows * Dsz / 4;
  cvt_f32_bf16<<<(unsigned)((n4 + 255) / 256), 256, 0, stream>>>(Z, Zh, n4);
  n4 = (long)D3 * Dsz / 4;
  cvt_f32_bf16<<<(unsigned)((n4 + 255) / 256), 256, 0, stream>>>(Wih, WihH, n4);
  cvt_f32_bf16<<<(unsigned)((n4 + 255) / 256), 256, 0, stream>>>(Whh, WhhH, n4);
  n4 = (long)Dsz * Dsz / 4;
  cvt_f32_bf16<<<(unsigned)((n4 + 255) / 256), 256, 0, stream>>>(Pw, PwH, n4);
  codebook_prep<<<Kc, 256, 0, stream>>>(cb, Ch, cn);

  // 1) quantize (+ indices, Q output, VQ mse accumulation)
  quantize_kernel<<<Nrows / 16, 256, 0, stream>>>(Z, cb, Zh, Ch, cn, Qout, idxF,
                                                  idxI, &sc[0]);

  // 2) GiC = codebook(bf16) @ Wih^T + bih (256x3072 replaces the 32768x3072 GEMM)
  gemm_abt<float><<<dim3(Kc / 16, D3 / 512), 256, 0, stream>>>(Ch, WihH, bih, GiC,
                                                               Kc, D3, Dsz);

  // 3) persistent GRU scan (8 co-resident blocks, spin grid barrier per step)
  hipFuncSetAttribute((const void*)gru_scan,
                      hipFuncAttributeMaxDynamicSharedMemorySize,
                      (Bsz * Dsz + Bsz * 128) * 2);
  gru_scan<<<8, 256, (Bsz * Dsz + Bsz * 128) * 2, stream>>>(
      GiC, idxI, WhhH, bhh, hb0, hb1, Hh, cnt, gen);

  // 4) projection Hp = Hh @ Pw^T + pb (bf16 output feeds the loss GEMMs)
  gemm_abt<bf16><<<dim3(Nrows / 16, Dsz / 512), 256, 0, stream>>>(Hh, PwH, pb, Hp,
                                                                  Nrows, Dsz, Dsz);

  // 5) CPC loss over all (k, t): 511 + 510 + 509 = 1530 blocks
  ctx_loss<<<1530, 256, 0, stream>>>(Hp, Zh, &sc[1], &sc[2]);

  // 6) scalars
  finalize<<<1, 32, 0, stream>>>(sc, outS);
}
